// TopologicalEmbedding_51316269252742
// MI455X (gfx1250) — matrix operand order repaired and checked by reading, not compiled, not run on hardware
//
#include <hip/hip_runtime.h>
#include <hip/hip_bf16.h>

typedef __attribute__((ext_vector_type(16))) _Float16 v16h;
typedef __attribute__((ext_vector_type(8)))  _Float16 v8h;
typedef __attribute__((ext_vector_type(8)))  float    v8f;

#define B_   128
#define P_   2048
#define F_   256
#define H_   1024
#define H4_  256
#define H2_  512
#define RES_ 4096

// workspace layout (bytes)
#define WS_PDSUM  0                                   // B*256 f32 pooled-sum accumulators
#define WS_PI     (WS_PDSUM + B_*H4_*4)               // B*8 f32 (E, O, x0, x4094, x4095)
#define WS_W2FRAG (WS_PI + B_*8*4)                    // 8*16*32*16 f16 (W2 B-fragments)
#define WS_WOFRAG (WS_W2FRAG + 8*16*32*16*2)          // 32*64*32*16 f16 (Wo B-fragments)
#define WS_COMB   (WS_WOFRAG + 32*64*32*16*2)         // B*1024 f16 (combined features)

__global__ void k_zero(float* p, int n) {
    int i = blockIdx.x * 256 + threadIdx.x;
    if (i < n) p[i] = 0.f;
}

// Pack fp32 row-major [K x N] weight into f16 WMMA B-fragment order:
// dst[((kb*NT + nt)*32 + lane)*16 + j], lane 0-15 -> col, K = kb*32 + (lane>=16)*8 + (j&7) + (j>=8)*16
__global__ void k_pack_frag(const float* __restrict__ src, _Float16* __restrict__ dst,
                            int KB, int NT, int N) {
    int tid = blockIdx.x * 256 + threadIdx.x;
    int j    = tid & 15;
    int lane = (tid >> 4) & 31;
    int nt   = (tid >> 9) % NT;
    int kb   = tid / (512 * NT);
    if (kb >= KB) return;
    int k = kb * 32 + ((lane >> 4) & 1) * 8 + (j & 7) + (j >> 3) * 16;
    int n = nt * 16 + (lane & 15);
    dst[tid] = (_Float16)src[k * N + n];
}

// Fused PD branch: Linear(3->256) + LayerNorm + exact GELU (VALU), stage f16 in LDS,
// Linear(256->256) via v_wmma_f32_16x16x32_f16 (2 row-tiles per wave, batched B loads),
// masked pooled-sum via atomics.
__global__ __launch_bounds__(128) void k_pd(const float* __restrict__ pd,
                                            const int*   __restrict__ lens,
                                            const float* __restrict__ W1,
                                            const float* __restrict__ b1,
                                            const float* __restrict__ lng,
                                            const float* __restrict__ lnb,
                                            const _Float16* __restrict__ w2f,
                                            float* __restrict__ pdsum) {
    extern __shared__ _Float16 sh[];           // 64 KB: 4 waves x 32 rows x 256 f16
    const int t    = threadIdx.x;
    const int lane = t & 31;
    const int wave = t >> 5;
    const int b    = blockIdx.x >> 4;          // 16 row-blocks per batch
    const int p0   = (blockIdx.x & 15) * 128;  // 128 rows per block
    const int pbase = p0 + wave * 32;          // this wave's 32 rows
    const int len  = lens[b];
    _Float16* slice = sh + wave * 32 * F_;

    // hoist per-feature params: feature f = lane + 32*j
    float w1a[8], w1b[8], w1c[8], bb[8], gg[8], be[8];
    #pragma unroll
    for (int j = 0; j < 8; j++) {
        int f = lane + 32 * j;
        w1a[j] = W1[f];  w1b[j] = W1[F_ + f];  w1c[j] = W1[2 * F_ + f];
        bb[j] = b1[f];   gg[j] = lng[f];       be[j] = lnb[f];
    }

    for (int r = 0; r < 32; r++) {
        const float* row = pd + (size_t)(b * P_ + pbase + r) * 3;
        float x0 = row[0], x1 = row[1], x2 = row[2];
        float y[8]; float s = 0.f, sq = 0.f;
        #pragma unroll
        for (int j = 0; j < 8; j++) {
            float v = fmaf(x0, w1a[j], fmaf(x1, w1b[j], fmaf(x2, w1c[j], bb[j])));
            y[j] = v; s += v; sq += v * v;
        }
        #pragma unroll
        for (int off = 16; off >= 1; off >>= 1) {
            s  += __shfl_xor(s,  off, 32);
            sq += __shfl_xor(sq, off, 32);
        }
        float mean = s * (1.f / 256.f);
        float var  = sq * (1.f / 256.f) - mean * mean;
        float rs   = rsqrtf(var + 1e-5f);
        #pragma unroll
        for (int j = 0; j < 8; j++) {
            float v = (y[j] - mean) * rs * gg[j] + be[j];
            float g = 0.5f * v * (1.f + erff(v * 0.70710678118f));   // exact GELU
            slice[r * F_ + lane + 32 * j] = (_Float16)g;
        }
    }
    __syncthreads();

    // Two 16x256 A-fragment sets (rows 0-15 and 16-31 of this wave's tile)
    const int hi8 = (lane >= 16) ? 8 : 0;
    const int m   = lane & 15;
    v16h a0[8], a1[8];
    #pragma unroll
    for (int kb = 0; kb < 8; kb++) {
        const v8h lo0 = *(const v8h*)(slice + m * F_ + kb * 32 + hi8);
        const v8h hv0 = *(const v8h*)(slice + m * F_ + kb * 32 + hi8 + 16);
        const v8h lo1 = *(const v8h*)(slice + (16 + m) * F_ + kb * 32 + hi8);
        const v8h hv1 = *(const v8h*)(slice + (16 + m) * F_ + kb * 32 + hi8 + 16);
        #pragma unroll
        for (int j = 0; j < 8; j++) {
            a0[kb][j] = lo0[j]; a0[kb][j + 8] = hv0[j];
            a1[kb][j] = lo1[j]; a1[kb][j + 8] = hv1[j];
        }
    }

    for (int nt = 0; nt < 16; nt++) {
        // batch all 8 k-step B fragments (16 independent b128 loads -> one clause)
        v16h bf[8];
        #pragma unroll
        for (int kb = 0; kb < 8; kb++)
            bf[kb] = *(const v16h*)(w2f + (((kb * 16 + nt) * 32 + lane) << 4));
        v8f c0 = {}, c1 = {};
        #pragma unroll
        for (int kb = 0; kb < 8; kb++) {
            c0 = __builtin_amdgcn_wmma_f32_16x16x32_f16(false, a0[kb], false, bf[kb],
                                                        (short)0, c0, false, false);
            c1 = __builtin_amdgcn_wmma_f32_16x16x32_f16(false, a1[kb], false, bf[kb],
                                                        (short)0, c1, false, false);
        }
        // masked reduction: both tiles feed the same output column -> merge first
        float pa = 0.f, pb = 0.f;
        #pragma unroll
        for (int j = 0; j < 8; j++) {
            int pA = pbase + hi8 + j;          // D layout: row = j + (lane>=16)*8
            int pB = pbase + 16 + hi8 + j;
            pa += (pA < len) ? c0[j] : 0.f;
            pb += (pB < len) ? c1[j] : 0.f;
        }
        pa += pb;
        pa += __shfl_xor(pa, 16, 32);          // fold lane / lane+16 (same column)
        if (lane < 16)
            unsafeAtomicAdd(&pdsum[b * H4_ + nt * 16 + lane], pa);
    }
}

// Persistence-image branch reduced to even/odd sums + edge samples.
__global__ void k_pi(const float* __restrict__ img, float* __restrict__ wspi) {
    __shared__ float red[256];
    int b = blockIdx.x, t = threadIdx.x;
    const float* x = img + (size_t)b * RES_;
    float s = 0.f;
    for (int i = t; i < RES_; i += 256) s += x[i];   // stride 256 keeps parity of t
    red[t] = s;
    __syncthreads();
    for (int off = 128; off >= 2; off >>= 1) {       // even offsets preserve parity
        if (t < off) red[t] += red[t + off];
        __syncthreads();
    }
    if (t == 0) {
        wspi[b * 8 + 0] = red[0];        // E = sum over even indices
        wspi[b * 8 + 1] = red[1];        // O = sum over odd indices
        wspi[b * 8 + 2] = x[0];
        wspi[b * 8 + 3] = x[RES_ - 2];
        wspi[b * 8 + 4] = x[RES_ - 1];
    }
}

// Build combined[B,1024] as f16: pd_pooled | betti_mean | pi_feat (5-tap conv collapse)
__global__ void k_combine(const float* __restrict__ pdsum, const int* __restrict__ lens,
                          const float* __restrict__ b2, const int* __restrict__ betti,
                          const float* __restrict__ btab, const float* __restrict__ convw,
                          const float* __restrict__ convb, const float* __restrict__ wspi,
                          _Float16* __restrict__ comb) {
    int b = blockIdx.x, t = threadIdx.x;
    float E = wspi[b * 8 + 0], O = wspi[b * 8 + 1];
    float xa = wspi[b * 8 + 2], xb = wspi[b * 8 + 3], xc = wspi[b * 8 + 4];
    float s0 = E - xb, s1 = O - xc, s2 = E, s3 = O, s4 = E - xa;
    int len = lens[b];
    float inv = (len > 0) ? 1.f / (float)len : 0.f;
    int i0 = min(max(betti[b * 3 + 0], 0), 9);
    int i1 = min(max(betti[b * 3 + 1], 0), 9);
    int i2 = min(max(betti[b * 3 + 2], 0), 9);
    #pragma unroll
    for (int q = 0; q < 4; q++) {
        int d = t + q * 256;
        float v;
        if (d < 256) {
            v = (len > 0) ? pdsum[b * 256 + d] * inv + b2[d] : 0.f;
        } else if (d < 512) {
            int e = d - 256;
            v = (btab[i0 * 256 + e] + btab[i1 * 256 + e] + btab[i2 * 256 + e]) * (1.f / 3.f);
        } else {
            int c = d - 512;
            const float* w = convw + c * 5;
            v = (w[0]*s0 + w[1]*s1 + w[2]*s2 + w[3]*s3 + w[4]*s4) * (1.f / 2048.f) + convb[c];
        }
        comb[(size_t)b * H_ + d] = (_Float16)v;
    }
}

// Output projection: [128 x 1024] @ [1024 x 1024] via WMMA, one 16x16 tile per wave.
__global__ __launch_bounds__(256) void k_out(const _Float16* __restrict__ comb,
                                             const _Float16* __restrict__ wof,
                                             const float* __restrict__ bo,
                                             float* __restrict__ out) {
    const int t = threadIdx.x, lane = t & 31, wave = t >> 5;
    const int tile = blockIdx.x * 8 + wave;   // 512 tiles = 8 row-tiles x 64 n-tiles
    const int rt = tile >> 6;
    const int nt = tile & 63;
    const int hi8 = (lane >= 16) ? 8 : 0;
    const int m = lane & 15;
    const _Float16* arow = comb + (size_t)(rt * 16 + m) * H_;
    v8f c = {};
    for (int kq = 0; kq < 8; kq++) {
        // batch 4 k-steps of A and B loads, then 4 back-to-back WMMAs
        v16h a[4], bf[4];
        #pragma unroll
        for (int u = 0; u < 4; u++) {
            int kb = kq * 4 + u;
            const v8h lo = *(const v8h*)(arow + kb * 32 + hi8);
            const v8h hv = *(const v8h*)(arow + kb * 32 + hi8 + 16);
            #pragma unroll
            for (int j = 0; j < 8; j++) { a[u][j] = lo[j]; a[u][j + 8] = hv[j]; }
            bf[u] = *(const v16h*)(wof + (((size_t)(kb * 64 + nt) * 32 + lane) << 4));
        }
        #pragma unroll
        for (int u = 0; u < 4; u++)
            c = __builtin_amdgcn_wmma_f32_16x16x32_f16(false, a[u], false, bf[u],
                                                       (short)0, c, false, false);
    }
    const int col = nt * 16 + (lane & 15);
    const float bb = bo[col];
    #pragma unroll
    for (int j = 0; j < 8; j++) {
        int row = rt * 16 + hi8 + j;
        out[(size_t)row * H_ + col] = c[j] + bb;
    }
}

extern "C" void kernel_launch(void* const* d_in, const int* in_sizes, int n_in,
                              void* d_out, int out_size, void* d_ws, size_t ws_size,
                              hipStream_t stream) {
    const float* pd    = (const float*)d_in[0];
    const int*   lens  = (const int*)  d_in[1];
    const int*   betti = (const int*)  d_in[2];
    const float* img   = (const float*)d_in[3];
    const float* W1    = (const float*)d_in[4];
    const float* b1    = (const float*)d_in[5];
    const float* lng   = (const float*)d_in[6];
    const float* lnb   = (const float*)d_in[7];
    const float* W2    = (const float*)d_in[8];
    const float* b2    = (const float*)d_in[9];
    const float* btab  = (const float*)d_in[10];
    const float* convw = (const float*)d_in[11];
    const float* convb = (const float*)d_in[12];
    const float* Wo    = (const float*)d_in[13];
    const float* bo    = (const float*)d_in[14];
    float* out = (float*)d_out;

    char* ws = (char*)d_ws;
    float*    pdsum = (float*)(ws + WS_PDSUM);
    float*    wspi  = (float*)(ws + WS_PI);
    _Float16* w2f   = (_Float16*)(ws + WS_W2FRAG);
    _Float16* wof   = (_Float16*)(ws + WS_WOFRAG);
    _Float16* comb  = (_Float16*)(ws + WS_COMB);

    // 1) zero accumulators (pdsum + pi region are contiguous)
    int nz = B_ * H4_ + B_ * 8;
    k_zero<<<(nz + 255) / 256, 256, 0, stream>>>(pdsum, nz);
    // 2) pack W2 / Wo into f16 WMMA B-fragment layout
    k_pack_frag<<<(8 * 16 * 512) / 256, 256, 0, stream>>>(W2, w2f, 8, 16, 256);
    k_pack_frag<<<(32 * 64 * 512) / 256, 256, 0, stream>>>(Wo, wof, 32, 64, 1024);
    // 3) fused PD branch (dominant 34 GFLOP GEMM on WMMA), 64 KB dynamic LDS
    k_pd<<<B_ * (P_ / 128), 128, 4 * 32 * F_ * 2, stream>>>(pd, lens, W1, b1, lng, lnb,
                                                           w2f, pdsum);
    // 4) persistence-image reduction
    k_pi<<<B_, 256, 0, stream>>>(img, wspi);
    // 5) combine branches into f16 features
    k_combine<<<B_, 256, 0, stream>>>(pdsum, lens, b2, betti, btab, convw, convb, wspi, comb);
    // 6) output projection on WMMA
    k_out<<<64, 256, 0, stream>>>(comb, wof, bo, out);
}
